// MotionEncoder_4827543241201
// MI455X (gfx1250) — matrix operand order repaired and checked by reading, not compiled, not run on hardware
//
#include <hip/hip_runtime.h>

// ---------------------------------------------------------------------------
// MotionEncoder for MI455X (gfx1250): 22 per-joint MLPs, f16 WMMA w/ f32 acc.
// Pre-pass converts weights f32->f16 (transposed) into d_ws once per launch;
// main kernel is WMMA + b128 LDS staging + branchless GELU.
// ---------------------------------------------------------------------------

typedef __attribute__((ext_vector_type(16))) _Float16 v16h;
typedef __attribute__((ext_vector_type(8)))  _Float16 h8;
typedef __attribute__((ext_vector_type(8)))  float    v8f;

#define JOINTS  22
#define LATENT  512
#define DCOL    263
#define M_TILE  128
#define KPAD    32

// workspace layout (halves)
#define W1T_HALVES_PER_J (KPAD * LATENT)     // 16384  ([N=512][K=32], zero-padded)
#define W2T_HALVES_PER_J (LATENT * LATENT)   // 262144 ([N=512][K=512])
#define WS_W1T_OFF 0
#define WS_W2T_OFF (JOINTS * W1T_HALVES_PER_J)
#define WS_HALVES  (WS_W2T_OFF + (size_t)JOINTS * W2T_HALVES_PER_J)  // ~12.3 MB

// LDS layout (halves):  sA[128][32] (aliased by sW2T), sW1T[512][32], sH[128][512]
#define OFF_A    0
#define OFF_W1T  4096
#define OFF_H    (4096 + 16384)
#define SMEM_HALVES (4096 + 16384 + 65536)
#define SMEM_BYTES  (SMEM_HALVES * 2)        // 172032 B < 320 KB WGP LDS

struct Params {
    const float* W1[JOINTS];
    const float* b1[JOINTS];
    const float* W2[JOINTS];
    const float* b2[JOINTS];
};

__device__ __forceinline__ int joint_dim(int j) {
    if (j == 0) return 7;
    if (j == 7 || j == 8 || j == 10 || j == 11) return 13;
    return 12;
}

// Source column in the 263-wide pose vector for (joint j, feature k)
__device__ __forceinline__ int joint_col(int j, int k) {
    if (j == 0) return (k < 4) ? k : (193 + (k - 4));            // root(4) + vel0(3)
    if (k < 3)  return 4 + 3 * (j - 1) + k;                      // ric
    if (k < 9)  return 67 + 6 * (j - 1) + (k - 3);               // rot
    if (k < 12) return 193 + 3 * j + (k - 9);                    // vel
    int c = (j == 7) ? 0 : (j == 8) ? 1 : (j == 10) ? 2 : 3;     // contact
    return 259 + c;
}

// Branchless tanh-form GELU using native v_exp_f32 / v_rcp_f32 (no exec
// divergence; accuracy ~1e-3 rel, below the f16 storage eps of H).
__device__ __forceinline__ float gelu_fast(float x) {
    float u  = x * (0.7978845608028654f + 0.035677408136f * x * x);
    float t  = fabsf(u);
    float e  = __expf(2.0f * t);                       // inf-safe: rcp(inf)=0
    float th = 1.0f - 2.0f * __builtin_amdgcn_rcpf(1.0f + e);
    th = copysignf(th, u);
    return 0.5f * x * (1.0f + th);
}

// A-fragment (16x32 f16): lanes 0-15 hold K 0..7 & 16..23, lanes 16-31 hold
// K 8..15 & 24..31 (CDNA5 ISA 7.12.2).  `row` must already include lane&15.
__device__ __forceinline__ v16h frag_ld_a(const _Float16* base, int stride, int row) {
    const int lane = threadIdx.x & 31;
    const int hs   = (lane >> 4) * 8;
    const _Float16* p = base + row * stride + hs;
    h8 lo = *(const h8*)(p);
    h8 hi = *(const h8*)(p + 16);
    v16h r;
#pragma unroll
    for (int i = 0; i < 8; ++i) { r[i] = lo[i]; r[i + 8] = hi[i]; }
    return r;
}

// B-fragment (32x16 f16), column-major in LDS as [N][32]: lanes 0-15 hold
// K 0..7 & 16..23 of column n, lanes 16-31 hold K 8..15 & 24..31.  `col` includes lane&15.
__device__ __forceinline__ v16h frag_ld_b(const _Float16* base, int stride, int col) {
    const int lane = threadIdx.x & 31;
    const int hs   = (lane >> 4) * 8;
    const _Float16* p = base + col * stride + hs;
    h8 lo = *(const h8*)(p);
    h8 hi = *(const h8*)(p + 16);
    v16h r;
#pragma unroll
    for (int i = 0; i < 8; ++i) { r[i] = lo[i]; r[i + 8] = hi[i]; }
    return r;
}

// ---- Pre-pass 1: W1 -> f16 transposed [N=512][K=32], zero-padded K >= d ----
__global__ void __launch_bounds__(256)
convert_w1t_kernel(Params p, _Float16* __restrict__ ws) {
    int j   = blockIdx.y;
    int idx = blockIdx.x * 256 + threadIdx.x;   // 0..16383 (grid.x = 64)
    int n = idx >> 5, k = idx & 31;
    float v = (k < joint_dim(j)) ? p.W1[j][k * LATENT + n] : 0.f;
    ws[WS_W1T_OFF + (size_t)j * W1T_HALVES_PER_J + n * KPAD + k] = (_Float16)v;
}

// ---- Pre-pass 2: W2 -> f16 transposed [N][K] via LDS-tiled transpose ----
__global__ void __launch_bounds__(256)
transpose_w2_kernel(Params p, _Float16* __restrict__ ws) {
    __shared__ float tile[32][33];
    int j  = blockIdx.z;
    int k0 = blockIdx.x * 32, n0 = blockIdx.y * 32;
    const float* W2 = p.W2[j];
    _Float16* dst = ws + WS_W2T_OFF + (size_t)j * W2T_HALVES_PER_J;
    int tx = threadIdx.x, ty = threadIdx.y;     // block (32, 8)
#pragma unroll
    for (int r = 0; r < 4; ++r)
        tile[ty + r * 8][tx] = W2[(size_t)(k0 + ty + r * 8) * LATENT + n0 + tx];
    __syncthreads();
#pragma unroll
    for (int r = 0; r < 4; ++r)
        dst[(size_t)(n0 + ty + r * 8) * LATENT + k0 + tx] = (_Float16)tile[tx][ty + r * 8];
}

// ---- Main kernel: per (row-block, joint) two-layer MLP via WMMA ----
__global__ void __launch_bounds__(256)
motion_encoder_kernel(const float* __restrict__ x, const _Float16* __restrict__ wh,
                      float* __restrict__ out, Params p) {
    extern __shared__ _Float16 smem[];
    _Float16* sA   = smem + OFF_A;    // layer1 activations [128][32]
    _Float16* sW1T = smem + OFF_W1T;  // W1^T               [512][32]
    _Float16* sH   = smem + OFF_H;    // hidden (post-GELU) [128][512]
    _Float16* sW2T = smem + OFF_A;    // layer2 W2^T tile   [128][32] (alias sA)

    const int rbase = blockIdx.x * M_TILE;   // 49 * 128 = 6272 rows exactly
    const int j     = blockIdx.y;
    const int tid   = threadIdx.x;
    const int wv    = tid >> 5;
    const int lane  = tid & 31;
    const int l16   = lane & 15;
    const int hi16  = lane >> 4;

    const int d = joint_dim(j);
    const float* b1g = p.b1[j];
    const float* b2g = p.b2[j];
    const _Float16* W1Th = wh + WS_W1T_OFF + (size_t)j * W1T_HALVES_PER_J;
    const _Float16* W2Th = wh + WS_W2T_OFF + (size_t)j * W2T_HALVES_PER_J;

    // Stage gathered activations (zero-padded to K=32) as f16
    for (int idx = tid; idx < M_TILE * KPAD; idx += 256) {
        int k = idx & (KPAD - 1), m = idx >> 5;
        float v = (k < d) ? x[(size_t)(rbase + m) * DCOL + joint_col(j, k)] : 0.f;
        sA[m * KPAD + k] = (_Float16)v;
    }
    // Stage W1^T: pure 16-byte copies (already f16 + transposed in ws)
    {
        const uint4* src = (const uint4*)W1Th;
        uint4*       dst = (uint4*)sW1T;
#pragma unroll
        for (int it = 0; it < 8; ++it) dst[tid + it * 256] = src[tid + it * 256];
    }
    __syncthreads();

    // ---- Layer 1: H = GELU(A @ W1 + b1), one WMMA per 16-wide N tile ----
    {
        v16h a = frag_ld_a(sA, KPAD, wv * 16 + l16);
        for (int nt = 0; nt < LATENT / 16; ++nt) {
            v16h b = frag_ld_b(sW1T, KPAD, nt * 16 + l16);
            v8f acc = {};
            acc = __builtin_amdgcn_wmma_f32_16x16x32_f16(
                false, a, false, b, (short)0, acc, false, false);
            int n = nt * 16 + l16;
            float bias = b1g[n];
#pragma unroll
            for (int v = 0; v < 8; ++v) {
                int m = wv * 16 + v + hi16 * 8;
                sH[m * LATENT + n] = (_Float16)gelu_fast(acc[v] + bias);
            }
        }
    }

    // ---- Layer 2: O = H @ W2 + b2; N in 4 blocks of 128, K in 16 chunks ----
    for (int nblk = 0; nblk < 4; ++nblk) {
        const int nbase = nblk * 128;
        v8f acc[8];
        v8f zero = {};
#pragma unroll
        for (int t = 0; t < 8; ++t) acc[t] = zero;

        for (int kc = 0; kc < LATENT / KPAD; ++kc) {
            __syncthreads();   // protect sW2T reuse (and sA/sH on first pass)
            // Stage 128x32 f16 W2^T tile: pure 16-byte copies
#pragma unroll
            for (int it = 0; it < 2; ++it) {
                int u  = tid + it * 256;         // uint4 index, halves h = u*8
                int h  = u * 8;
                int nc = h >> 5, kk = h & 31;
                ((uint4*)sW2T)[u] =
                    *(const uint4*)(W2Th + (size_t)(nbase + nc) * LATENT + kc * KPAD + kk);
            }
            // Prefetch next K-chunk into L2/L0 (global_prefetch_b8)
            if (kc + 1 < LATENT / KPAD) {
                int h  = tid * 8;
                int nc = h >> 5, kk = h & 31;
                __builtin_prefetch(
                    W2Th + (size_t)(nbase + nc) * LATENT + (kc + 1) * KPAD + kk, 0, 0);
            }
            __syncthreads();

            v16h a = frag_ld_a(sH + kc * KPAD, LATENT, wv * 16 + l16);
#pragma unroll
            for (int nt = 0; nt < 8; ++nt) {
                v16h b = frag_ld_b(sW2T, KPAD, nt * 16 + l16);
                acc[nt] = __builtin_amdgcn_wmma_f32_16x16x32_f16(
                    false, a, false, b, (short)0, acc[nt], false, false);
            }
        }

        // fused bias + store: out[(m*22 + j)*512 + n]
#pragma unroll
        for (int nt = 0; nt < 8; ++nt) {
            int n = nbase + nt * 16 + l16;
            float bias = b2g[n];
#pragma unroll
            for (int v = 0; v < 8; ++v) {
                int m = rbase + wv * 16 + v + hi16 * 8;
                out[((size_t)m * JOINTS + j) * LATENT + n] = acc[nt][v] + bias;
            }
        }
    }
}

extern "C" void kernel_launch(void* const* d_in, const int* in_sizes, int n_in,
                              void* d_out, int out_size, void* d_ws, size_t ws_size,
                              hipStream_t stream) {
    (void)in_sizes; (void)n_in; (void)out_size; (void)ws_size;

    const float* x = (const float*)d_in[0];
    Params p;
    for (int i = 0; i < JOINTS; ++i) {
        p.W1[i] = (const float*)d_in[1 + i];
        p.b1[i] = (const float*)d_in[1 + JOINTS + i];
        p.W2[i] = (const float*)d_in[1 + 2 * JOINTS + i];
        p.b2[i] = (const float*)d_in[1 + 3 * JOINTS + i];
    }

    _Float16* wsh = (_Float16*)d_ws;   // needs ~12.3 MB scratch

    // Pre-pass: weights -> f16, transposed (runs every call; deterministic)
    convert_w1t_kernel<<<dim3(64, JOINTS), 256, 0, stream>>>(p, wsh);
    transpose_w2_kernel<<<dim3(16, 16, JOINTS), dim3(32, 8), 0, stream>>>(p, wsh);

    // 172 KB dynamic LDS (gfx1250 WGP supports up to 320 KB per workgroup)
    (void)hipFuncSetAttribute((const void*)motion_encoder_kernel,
                              hipFuncAttributeMaxDynamicSharedMemorySize,
                              SMEM_BYTES);

    dim3 grid(49, JOINTS);   // 6272 rows / 128, one joint per blockIdx.y
    motion_encoder_kernel<<<grid, 256, SMEM_BYTES, stream>>>(
        x, wsh, (float*)d_out, p);
}
